// AdditiveAttention_32461362823707
// MI455X (gfx1250) — compile-verified, matrix operand order checked by reading, not run on hardware
//
#include <hip/hip_runtime.h>

// MI455X / gfx1250: wave32, WMMA f32<-f16 16x16x32 (codegen-confirmed builtin).
typedef __attribute__((ext_vector_type(16))) _Float16 v16h;
typedef __attribute__((ext_vector_type(8)))  _Float16 h8;
typedef __attribute__((ext_vector_type(8)))  float    v8f;

#define D_MODEL 512
#define DK      64
#define LQ      1024
#define LK      1024
#define NB      2
#define LN_EPS  1e-6f

// 16-bit A/B fragment (ISA 7.12.2): lane holds two contiguous 8-half runs:
// K in [8*half, 8*half+8) and [16+8*half, 16+8*half+8).  Load each run as b128.
__device__ __forceinline__ v16h load_frag16(const _Float16* run0, const _Float16* run1) {
    h8 r0 = *(const h8*)run0;
    h8 r1 = *(const h8*)run1;
    v16h a;
    #pragma unroll
    for (int i = 0; i < 8; ++i) { a[i] = r0[i]; a[8 + i] = r1[i]; }
    return a;
}

// ---------------------------------------------------------------------------
// K1: projections. grid (512, 3), block (64, 4).
//   which=0: qp[2048][64] f32
//   which=1: kpT[b][64][1024] f32      (j-contiguous for coalesced score loop)
//   which=2: vpT_h[b][64][1024] f16    (k-contiguous rows -> b128 B-fragments)
// ---------------------------------------------------------------------------
__global__ void proj_kernel(const float* __restrict__ q, const float* __restrict__ k,
                            const float* __restrict__ v,
                            const float* __restrict__ Wq, const float* __restrict__ Wk,
                            const float* __restrict__ Wv,
                            float* __restrict__ qp, float* __restrict__ kpT,
                            _Float16* __restrict__ vpT_h) {
    const int which = blockIdx.y;
    const float* X = (which == 0) ? q : (which == 1) ? k : v;
    const float* W = (which == 0) ? Wq : (which == 1) ? Wk : Wv;

    __shared__ float xrow[4][D_MODEL];              // 8 KB
    const int col = threadIdx.x;                    // 0..63
    const int row = blockIdx.x * 4 + threadIdx.y;   // 0..2047
    const int tid = threadIdx.y * 64 + threadIdx.x;

    for (int idx = tid; idx < 4 * D_MODEL; idx += 256) {
        int r = idx >> 9, d = idx & (D_MODEL - 1);
        xrow[r][d] = X[(size_t)(blockIdx.x * 4 + r) * D_MODEL + d];
    }
    __syncthreads();

    float acc = 0.f;
    #pragma unroll 8
    for (int d = 0; d < D_MODEL; ++d)
        acc = fmaf(xrow[threadIdx.y][d], W[d * DK + col], acc);

    const int b = row / LK, j = row % LK;
    if (which == 0) {
        qp[(size_t)row * DK + col] = acc;
    } else if (which == 1) {
        kpT[(size_t)b * DK * LK + (size_t)col * LK + j] = acc;
    } else {
        vpT_h[((size_t)b * DK + col) * LK + j] = (_Float16)acc;
    }
}

// Tiny prep: WfcT_h[c][kk] = (f16) Wfc[kk][c]   (64x512 -> 512x64, k-contiguous)
__global__ void wfcT_kernel(const float* __restrict__ Wfc, _Float16* __restrict__ WfcT_h) {
    int idx = blockIdx.x * 256 + threadIdx.x;       // 0..32767
    int kk = idx >> 9, c = idx & (D_MODEL - 1);
    WfcT_h[(size_t)c * DK + kk] = (_Float16)Wfc[(size_t)kk * D_MODEL + c];
}

// ---------------------------------------------------------------------------
// K2: additive scores + softmax + attn@vp (WMMA, split-K across 8 waves).
// grid 128 blocks (batch x 16-row query tile), block 256 threads.
// ---------------------------------------------------------------------------
#define SCH_STRIDE 1032   // halves; stride*2B = 2064B -> banks m*4, conflict-free b128

__global__ void attn_kernel(const float* __restrict__ qp, const float* __restrict__ kpT,
                            const _Float16* __restrict__ vpT_h,
                            const float* __restrict__ vparam,
                            float* __restrict__ attn_out, _Float16* __restrict__ out_av_h) {
    __shared__ float sc[16 * LK];                                  // 64 KB raw scores
    __shared__ __align__(16) _Float16 sc_h[16 * SCH_STRIDE];       // 33 KB f16 attn strip
    __shared__ float qt[16][DK];                                   // 4 KB
    __shared__ float vpar[DK];
    __shared__ float part[2 * 16 * DK];                            // 8 KB split-K partials

    const int b     = blockIdx.x >> 6;
    const int qbase = (blockIdx.x & 63) * 16;
    const int tid   = threadIdx.x;        // 0..255
    const int wave  = tid >> 5;           // 0..7
    const int lane  = tid & 31;

    for (int idx = tid; idx < 16 * DK; idx += 256) {
        int i = idx >> 6, d = idx & 63;
        qt[i][d] = qp[(size_t)(b * LQ + qbase + i) * DK + d];
    }
    if (tid < DK) vpar[tid] = vparam[tid];
    __syncthreads();

    // ---- Phase A: scores = sum_d tanh(qp_i + kp_j) * v_param  (VALU/trans-bound)
    const size_t kofs = (size_t)b * DK * LK;
    for (int jb = 0; jb < LK / 256; ++jb) {
        int j = jb * 256 + tid;           // coalesced in kpT's j-major layout
        for (int i = 0; i < 16; ++i) {
            float acc = 0.f;
            #pragma unroll 8
            for (int d = 0; d < DK; ++d)
                acc = fmaf(tanhf(qt[i][d] + kpT[kofs + (size_t)d * LK + j]), vpar[d], acc);
            sc[i * LK + j] = acc;
        }
    }
    __syncthreads();

    // ---- Phase B: row softmax (wave32 shuffle), emit f32 attn + f16 strip
    for (int i = wave; i < 16; i += 8) {
        float mx = -3.402823466e38f;
        for (int j = lane; j < LK; j += 32) mx = fmaxf(mx, sc[i * LK + j]);
        #pragma unroll
        for (int off = 16; off; off >>= 1) mx = fmaxf(mx, __shfl_xor(mx, off, 32));
        float sum = 0.f;
        for (int j = lane; j < LK; j += 32) {
            float p = __expf(sc[i * LK + j] - mx);
            sc[i * LK + j] = p;
            sum += p;
        }
        #pragma unroll
        for (int off = 16; off; off >>= 1) sum += __shfl_xor(sum, off, 32);
        float inv = 1.f / sum;
        for (int j = lane; j < LK; j += 32) {
            float p = sc[i * LK + j] * inv;
            attn_out[(size_t)(b * LQ + qbase + i) * LK + j] = p;
            sc_h[i * SCH_STRIDE + j] = (_Float16)p;
        }
    }
    __syncthreads();

    // ---- Phase C: out_av(16x64) = attn @ vp.  8 waves: 4 col-tiles x 2 K-halves.
    const int half  = lane >> 4;
    const int mn    = lane & 15;
    const int ntile = wave & 3;
    const int khalf = wave >> 2;

    const _Float16* arow = sc_h + mn * SCH_STRIDE + half * 8;
    const _Float16* brow = vpT_h + ((size_t)b * DK + ntile * 16 + mn) * LK + half * 8;

    v8f c = {};
    #pragma unroll 4
    for (int kb = khalf * (LK / 2); kb < (khalf + 1) * (LK / 2); kb += 32) {
        v16h a  = load_frag16(arow + kb, arow + kb + 16);   // 2x ds_load_b128
        v16h bf = load_frag16(brow + kb, brow + kb + 16);   // 2x global_load_b128
        c = __builtin_amdgcn_wmma_f32_16x16x32_f16(false, a, false, bf,
                                                   (short)0, c, false, false);
    }
    #pragma unroll
    for (int g = 0; g < 8; ++g) {
        int m = g + half * 8;
        part[(khalf * 16 + m) * DK + ntile * 16 + mn] = c[g];
    }
    __syncthreads();

    for (int idx = tid; idx < 16 * DK; idx += 256) {
        out_av_h[(size_t)(b * LQ + qbase + (idx >> 6)) * DK + (idx & 63)] =
            (_Float16)(part[idx] + part[16 * DK + idx]);
    }
}

// ---------------------------------------------------------------------------
// K3: out = LN(out_av @ Wfc + residual). grid 128, block 128 (4 waves).
// Pure-f16 operand feeds: zero conversion VALU per WMMA.
// ---------------------------------------------------------------------------
#define OB_STRIDE 516

__global__ void fc_ln_kernel(const _Float16* __restrict__ out_av_h,
                             const _Float16* __restrict__ WfcT_h,
                             const float* __restrict__ q,
                             const float* __restrict__ gamma, const float* __restrict__ beta,
                             float* __restrict__ out) {
    __shared__ float ob[16 * OB_STRIDE];  // ~33 KB padded
    __shared__ float red[2][16][8];
    __shared__ float mu_s[16], rstd_s[16];

    const int b     = blockIdx.x >> 6;
    const int qbase = (blockIdx.x & 63) * 16;
    const int tid   = threadIdx.x;
    const int wave  = tid >> 5;
    const int lane  = tid & 31;
    const int half  = lane >> 4;
    const int mn    = lane & 15;

    // A fragments: rows qbase..+15, K=64 = two 16x32 chunks, four b128 loads
    const _Float16* arow = out_av_h + (size_t)(b * LQ + qbase + mn) * DK + half * 8;
    const v16h a0 = load_frag16(arow,      arow + 16);
    const v16h a1 = load_frag16(arow + 32, arow + 48);

    #pragma unroll
    for (int t = 0; t < 8; ++t) {
        const int ctile = wave * 8 + t;
        const _Float16* brow = WfcT_h + (size_t)(ctile * 16 + mn) * DK + half * 8;
        v16h b0 = load_frag16(brow,      brow + 16);
        v16h b1 = load_frag16(brow + 32, brow + 48);
        v8f c = {};
        c = __builtin_amdgcn_wmma_f32_16x16x32_f16(false, a0, false, b0, (short)0, c, false, false);
        c = __builtin_amdgcn_wmma_f32_16x16x32_f16(false, a1, false, b1, (short)0, c, false, false);
        #pragma unroll
        for (int g = 0; g < 8; ++g) {
            int m   = g + half * 8;
            int col = ctile * 16 + mn;
            ob[m * OB_STRIDE + col] =
                c[g] + q[(size_t)(b * LQ + qbase + m) * D_MODEL + col];  // residual
        }
    }
    __syncthreads();

    // LayerNorm over 512
    {
        int r = tid & 15, p = tid >> 4;
        float s1 = 0.f, s2 = 0.f;
        #pragma unroll 8
        for (int ci = 0; ci < 64; ++ci) {
            float x = ob[r * OB_STRIDE + p * 64 + ci];
            s1 += x; s2 += x * x;
        }
        red[0][r][p] = s1; red[1][r][p] = s2;
    }
    __syncthreads();
    if (tid < 16) {
        float sa = 0.f, sb = 0.f;
        #pragma unroll
        for (int p = 0; p < 8; ++p) { sa += red[0][tid][p]; sb += red[1][tid][p]; }
        float mu  = sa * (1.0f / D_MODEL);
        float var = sb * (1.0f / D_MODEL) - mu * mu;
        mu_s[tid]   = mu;
        rstd_s[tid] = rsqrtf(var + LN_EPS);
    }
    __syncthreads();
    for (int idx = tid; idx < 16 * D_MODEL; idx += 128) {
        int r = idx >> 9, ccol = idx & (D_MODEL - 1);
        float x = (ob[r * OB_STRIDE + ccol] - mu_s[r]) * rstd_s[r] * gamma[ccol] + beta[ccol];
        out[(size_t)(b * LQ + qbase + r) * D_MODEL + ccol] = x;
    }
}

// ---------------------------------------------------------------------------
extern "C" void kernel_launch(void* const* d_in, const int* in_sizes, int n_in,
                              void* d_out, int out_size, void* d_ws, size_t ws_size,
                              hipStream_t stream) {
    const float* q      = (const float*)d_in[0];
    const float* k      = (const float*)d_in[1];
    const float* v      = (const float*)d_in[2];
    const float* Wq     = (const float*)d_in[3];
    const float* Wk     = (const float*)d_in[4];
    const float* Wv     = (const float*)d_in[5];
    const float* vparam = (const float*)d_in[6];
    const float* Wfc    = (const float*)d_in[7];
    const float* gamma  = (const float*)d_in[8];
    const float* beta   = (const float*)d_in[9];

    // ws: qp(f32 131072) | kpT(f32 131072) | f16 region: vpT | out_av | WfcT
    float*    ws       = (float*)d_ws;
    float*    qp       = ws;
    float*    kpT      = ws + 131072;
    _Float16* hbase    = (_Float16*)(ws + 262144);
    _Float16* vpT_h    = hbase;             // 131072 halves
    _Float16* out_av_h = hbase + 131072;    // 131072 halves
    _Float16* WfcT_h   = hbase + 262144;    //  32768 halves

    float* out  = (float*)d_out;                        // [2,1024,512]
    float* attn = out + (size_t)NB * LQ * D_MODEL;      // [2,1024,1024]

    proj_kernel <<<dim3(512, 3), dim3(64, 4), 0, stream>>>(q, k, v, Wq, Wk, Wv,
                                                           qp, kpT, vpT_h);
    wfcT_kernel <<<dim3(128),    dim3(256),  0, stream>>>(Wfc, WfcT_h);
    attn_kernel <<<dim3(128),    dim3(256),  0, stream>>>(qp, kpT, vpT_h, vparam,
                                                          attn, out_av_h);
    fc_ln_kernel<<<dim3(128),    dim3(128),  0, stream>>>(out_av_h, WfcT_h, q,
                                                          gamma, beta, out);
}
